// SingleHeadCausalSelfAttention_29446295781399
// MI455X (gfx1250) — compile-verified, matrix operand order checked by reading, not run on hardware
//
#include <hip/hip_runtime.h>
#include <hip/hip_bf16.h>
#include <cstdint>

// Problem constants (match reference)
#define BB 8
#define SS 2048
#define HH 1024

typedef __attribute__((ext_vector_type(16))) _Float16 v16h;
typedef __attribute__((ext_vector_type(8)))  _Float16 v8h;
typedef __attribute__((ext_vector_type(8)))  float    v8f;

#define LDS_PD 40                    // padded row stride (halfs): conflict-free, 16B aligned
#define A_PART (128 * LDS_PD)        // A tile: 128 rows x 32 K (padded), halfs
#define B_PART (64 * LDS_PD)         // B tile: 64 rows x 32 K (padded), halfs
#define STAGE_TOT (A_PART + B_PART)  // one pipeline stage, halfs (7680)

// ---------------------------------------------------------------------------
// WMMA helper: D = A(16x32 f16) x B(32x16 f16) + C(16x16 f32)
// ---------------------------------------------------------------------------
static __device__ __forceinline__ v8f wmma16(v16h a, v16h b, v8f c) {
  return __builtin_amdgcn_wmma_f32_16x16x32_f16(
      /*neg_a=*/false, a, /*neg_b=*/false, b,
      /*c_mod=*/(short)0, c, /*reuse_a=*/false, /*reuse_b=*/false);
}

static __device__ __forceinline__ v16h cat16(v8h a, v8h b) {
  return __builtin_shufflevector(a, b, 0,1,2,3,4,5,6,7,8,9,10,11,12,13,14,15);
}

// CDNA5 async global->LDS copy (16B per lane), tracked by ASYNCcnt.
// lds_byte = wave-relative LDS byte address (low 32 bits of generic pointer).
static __device__ __forceinline__ void async_copy_b128(uint32_t lds_byte,
                                                       const void* gaddr) {
  asm volatile("global_load_async_to_lds_b128 %0, %1, off"
               :: "v"(lds_byte), "v"(gaddr)
               : "memory");
}

// Refill one stage: A tile (2 x b128/thread) + B tile (1 x b128/thread).
static __device__ __forceinline__ void refill_stage(
    uint32_t stByte, int k,
    const _Float16* gA0, const _Float16* gA1, const _Float16* gB,
    uint32_t slotA0, uint32_t slotA1, uint32_t slotB) {
  async_copy_b128(stByte + slotA0, gA0 + k);
  async_copy_b128(stByte + slotA1, gA1 + k);
  async_copy_b128(stByte + slotB,  gB  + k);
}

// ---------------------------------------------------------------------------
// One pipeline step: wait for the stage staged two steps ago (allow the one
// in-flight refill = 3 async ops), barrier, async-refill the stage read one
// step ago, then 4 WMMAs fed entirely from LDS (A: 2 ds_loads, B: 8).
// ---------------------------------------------------------------------------
static __device__ __forceinline__ void gemm_step(
    int k0, int kend, uint32_t refillByte, const _Float16* pcS,
    const _Float16* gA0, const _Float16* gA1, const _Float16* gB,
    uint32_t slotA0, uint32_t slotA1, uint32_t slotB,
    int wid, int r, int hi, v8f acc[4]) {
  if (k0 + 32 < kend) asm volatile("s_wait_asynccnt 0x3" ::: "memory");
  else                asm volatile("s_wait_asynccnt 0x0" ::: "memory");
  __syncthreads();

  if (k0 + 64 < kend)
    refill_stage(refillByte, k0 + 64, gA0, gA1, gB, slotA0, slotA1, slotB);

  // A fragment for this wave's 16 rows (same layout as B fragments)
  const _Float16* ap = pcS + (wid * 16 + r) * LDS_PD + hi * 8;
  v8h a0 = *(const v8h*)(ap);
  v8h a1 = *(const v8h*)(ap + 16);

  v8h b0[4], b1[4];
#pragma unroll
  for (int t = 0; t < 4; ++t) {
    const _Float16* p = pcS + A_PART + (t * 16 + r) * LDS_PD + hi * 8;
    b0[t] = *(const v8h*)(p);
    b1[t] = *(const v8h*)(p + 16);
  }
  v16h av = cat16(a0, a1);
#pragma unroll
  for (int t = 0; t < 4; ++t)
    acc[t] = wmma16(av, cat16(b0[t], b1[t]), acc[t]);
}

// ---------------------------------------------------------------------------
// Block-cooperative GEMM core: triple-buffered, fully async-LDS pipeline,
// manually unrolled by the pipeline period (3) so all stage addresses are
// static. Loop-carried state is only the accumulators: no register rotation,
// no loadcnt waits, no WMMA-hazard nops.
//   Block tile 128x64; wave w computes rows m0blk + w*16.
// Fragment layout per CDNA5 ISA: lanes 0-15 row r get K [0..7,16..23],
// lanes 16-31 get K [8..15,24..31].
// ---------------------------------------------------------------------------
static __device__ __forceinline__ void gemm_core(
    const _Float16* __restrict__ A, int lda,
    const _Float16* __restrict__ Bsrc, int ldb,
    int m0blk, int n0, int kend,
    _Float16* lb, v8f acc[4]) {
  const int tid  = threadIdx.x;
  const int wid  = tid >> 5;
  const int lane = tid & 31;
  const int r    = lane & 15;
  const int hi   = lane >> 4;
  const int crow = tid >> 2;   // 0..63: copy row
  const int cch  = tid & 3;    // 0..3 : 8-half chunk within the 32-K tile

  // per-thread global copy pointers
  const _Float16* gB  = Bsrc + (size_t)(n0 + crow)         * (size_t)ldb + cch * 8;
  const _Float16* gA0 = A    + (size_t)(m0blk + crow)      * (size_t)lda + cch * 8;
  const _Float16* gA1 = A    + (size_t)(m0blk + 64 + crow) * (size_t)lda + cch * 8;

  // per-thread LDS byte slots within a stage
  const uint32_t slotA0 = (uint32_t)((crow * LDS_PD + cch * 8) * 2);
  const uint32_t slotA1 = slotA0 + (uint32_t)(64 * LDS_PD * 2);
  const uint32_t slotB  = (uint32_t)((A_PART + crow * LDS_PD + cch * 8) * 2);

  const uint32_t  base = (uint32_t)(uintptr_t)lb;
  const uint32_t  st0 = base;
  const uint32_t  st1 = base + STAGE_TOT * 2;
  const uint32_t  st2 = base + 2 * STAGE_TOT * 2;
  const _Float16* p0 = lb;
  const _Float16* p1 = lb + STAGE_TOT;
  const _Float16* p2 = lb + 2 * STAGE_TOT;

  // prologue: stage0 <- k=0; stage1 <- k=32
  refill_stage(st0, 0, gA0, gA1, gB, slotA0, slotA1, slotB);
  if (kend > 32)
    refill_stage(st1, 32, gA0, gA1, gB, slotA0, slotA1, slotB);

  // steady loop, unrolled by pipeline period 3 (step i: consume stage i%3,
  // refill stage (i+2)%3). Guards are block-uniform.
  for (int k0 = 0; k0 < kend; k0 += 96) {
    gemm_step(k0, kend, st2, p0, gA0, gA1, gB, slotA0, slotA1, slotB,
              wid, r, hi, acc);
    if (k0 + 32 < kend)
      gemm_step(k0 + 32, kend, st0, p1, gA0, gA1, gB, slotA0, slotA1, slotB,
                wid, r, hi, acc);
    if (k0 + 64 < kend)
      gemm_step(k0 + 64, kend, st1, p2, gA0, gA1, gB, slotA0, slotA1, slotB,
                wid, r, hi, acc);
  }
}

// ---------------------------------------------------------------------------
// f32 -> f16 conversion
// ---------------------------------------------------------------------------
__global__ void cvt_f32_f16_kernel(const float* __restrict__ src,
                                   _Float16* __restrict__ dst, int n) {
  int i = blockIdx.x * blockDim.x + threadIdx.x;
  if (i < n) dst[i] = (_Float16)src[i];
}

// ---------------------------------------------------------------------------
// Projection GEMM: C[M,HH] = A[M,HH](f16) * W[HH,HH](f16,[n,k])^T + bias
// mode 0: f16 row-major (Q,K) | mode 1: f16 transposed per batch (V->Vt)
// mode 2: f32 row-major (final output)
// Block tile 128x64 (8 waves x 16 rows).
// ---------------------------------------------------------------------------
__global__ void gemm_proj_kernel(const _Float16* __restrict__ A,
                                 const _Float16* __restrict__ W,
                                 const float* __restrict__ bias,
                                 _Float16* __restrict__ Ch,
                                 float* __restrict__ Cf,
                                 int mode) {
  __shared__ _Float16 lb[3 * STAGE_TOT];
  const int nblk = HH / 64; // 16
  int bm = blockIdx.x / nblk, bn = blockIdx.x % nblk;
  int m0blk = bm * 128;
  int m0 = m0blk + (threadIdx.x >> 5) * 16;
  int n0 = bn * 64;

  v8f acc[4] = {};
  gemm_core(A, HH, W, HH, m0blk, n0, HH, lb, acc);

  int lane = threadIdx.x & 31;
  int col  = lane & 15;
  int hi   = lane >> 4;
#pragma unroll
  for (int t = 0; t < 4; ++t) {
    int c = n0 + t * 16 + col;
    float bv = bias[c];
#pragma unroll
    for (int j = 0; j < 8; ++j) {
      int row = m0 + j + hi * 8;
      float v = acc[t][j] + bv;
      if (mode == 0) {
        Ch[(size_t)row * HH + c] = (_Float16)v;
      } else if (mode == 1) {
        int b = row / SS, s = row % SS;
        Ch[((size_t)b * HH + c) * SS + s] = (_Float16)v;
      } else {
        Cf[(size_t)row * HH + c] = v;
      }
    }
  }
}

// ---------------------------------------------------------------------------
// Scores GEMM: scores[b] = mask(Q[b] * K[b]^T / 32), f16 [S,S]
// Blocks entirely above the diagonal skip all compute.
// ---------------------------------------------------------------------------
__global__ void gemm_scores_kernel(const _Float16* __restrict__ Qh,
                                   const _Float16* __restrict__ Kh,
                                   const int* __restrict__ amask,
                                   _Float16* __restrict__ Sc) {
  __shared__ _Float16 lb[3 * STAGE_TOT];
  const int nblk = SS / 64; // 32
  int b  = blockIdx.y;
  int bm = blockIdx.x / nblk, bn = blockIdx.x % nblk;
  int m0blk = bm * 128;
  int m0 = m0blk + (threadIdx.x >> 5) * 16;
  int n0 = bn * 64;

  const _Float16* Q = Qh + (size_t)b * SS * HH;
  const _Float16* K = Kh + (size_t)b * SS * HH;
  _Float16* out = Sc + (size_t)b * SS * SS;

  int lane = threadIdx.x & 31;
  int col  = lane & 15;
  int hi   = lane >> 4;
  _Float16 ninf = (_Float16)(-__builtin_huge_valf());

  if (n0 > m0blk + 127) { // whole block above diagonal
#pragma unroll
    for (int t = 0; t < 4; ++t) {
      int c = n0 + t * 16 + col;
#pragma unroll
      for (int j = 0; j < 8; ++j)
        out[(size_t)(m0 + j + hi * 8) * SS + c] = ninf;
    }
    return;
  }

  v8f acc[4] = {};
  gemm_core(Q, HH, K, HH, m0blk, n0, HH, lb, acc);

  const float scale = 0.03125f; // 1/sqrt(1024)
#pragma unroll
  for (int t = 0; t < 4; ++t) {
    int c  = n0 + t * 16 + col;
    int mk = amask[b * SS + c];
#pragma unroll
    for (int j = 0; j < 8; ++j) {
      int row = m0 + j + hi * 8;
      bool dead = (c > row) || (mk == 0);
      float v = acc[t][j] * scale;
      out[(size_t)row * SS + c] = dead ? ninf : (_Float16)v;
    }
  }
}

// ---------------------------------------------------------------------------
// In-place row softmax over f16 scores: one block (256 thr) per row.
// ---------------------------------------------------------------------------
__global__ void softmax_kernel(_Float16* __restrict__ Sc) {
  __shared__ float red[256];
  _Float16* row = Sc + ((size_t)blockIdx.y * SS + blockIdx.x) * SS;
  int t = threadIdx.x;

  float v[8];
  float m = -__builtin_huge_valf();
#pragma unroll
  for (int j = 0; j < 8; ++j) {
    v[j] = (float)row[t + j * 256];
    m = fmaxf(m, v[j]);
  }
  red[t] = m;
  __syncthreads();
  for (int off = 128; off > 0; off >>= 1) {
    if (t < off) red[t] = fmaxf(red[t], red[t + off]);
    __syncthreads();
  }
  m = red[0];
  __syncthreads();

  float s = 0.f;
#pragma unroll
  for (int j = 0; j < 8; ++j) {
    v[j] = __expf(v[j] - m); // exp(-inf - m) = 0
    s += v[j];
  }
  red[t] = s;
  __syncthreads();
  for (int off = 128; off > 0; off >>= 1) {
    if (t < off) red[t] += red[t + off];
    __syncthreads();
  }
  float inv = 1.0f / red[0];
#pragma unroll
  for (int j = 0; j < 8; ++j)
    row[t + j * 256] = (_Float16)(v[j] * inv);
}

// ---------------------------------------------------------------------------
// PV GEMM: attnout[b] = probs[b] (S x S) * Vt[b]^T -> [S, H], f16.
// probs row q is exactly 0 past column q, so K-loop stops at m0blk+128.
// ---------------------------------------------------------------------------
__global__ void gemm_pv_kernel(const _Float16* __restrict__ Pr,
                               const _Float16* __restrict__ Vt,
                               _Float16* __restrict__ Ao) {
  __shared__ _Float16 lb[3 * STAGE_TOT];
  const int nblk = HH / 64; // 16
  int b  = blockIdx.y;
  int bm = blockIdx.x / nblk, bn = blockIdx.x % nblk;
  int m0blk = bm * 128;
  int m0 = m0blk + (threadIdx.x >> 5) * 16;
  int n0 = bn * 64;

  const _Float16* P = Pr + (size_t)b * SS * SS; // [S,S], ld=S
  const _Float16* V = Vt + (size_t)b * HH * SS; // [H,S], ld=S

  v8f acc[4] = {};
  gemm_core(P, SS, V, SS, m0blk, n0, m0blk + 128, lb, acc);

  int lane = threadIdx.x & 31;
  int col  = lane & 15;
  int hi   = lane >> 4;
#pragma unroll
  for (int t = 0; t < 4; ++t) {
    int c = n0 + t * 16 + col;
#pragma unroll
    for (int j = 0; j < 8; ++j) {
      int row = m0 + j + hi * 8;
      Ao[((size_t)b * SS + row) * HH + c] = (_Float16)acc[t][j];
    }
  }
}

// ---------------------------------------------------------------------------
// Launcher
// ---------------------------------------------------------------------------
extern "C" void kernel_launch(void* const* d_in, const int* in_sizes, int n_in,
                              void* d_out, int out_size, void* d_ws, size_t ws_size,
                              hipStream_t stream) {
  const float* x  = (const float*)d_in[0];
  const int*   am = (const int*)d_in[1];
  const float* wq = (const float*)d_in[2];
  const float* bq = (const float*)d_in[3];
  const float* wk = (const float*)d_in[4];
  const float* bk = (const float*)d_in[5];
  const float* wv = (const float*)d_in[6];
  const float* bv = (const float*)d_in[7];
  const float* wo = (const float*)d_in[8];
  const float* bo = (const float*)d_in[9];
  float* out = (float*)d_out;

  const int M  = BB * SS;        // 16384
  const int NX = M * HH;         // 16,777,216
  const int NW = HH * HH;        // 1,048,576

  // workspace layout (f16 element offsets)
  _Float16* wsh = (_Float16*)d_ws;
  _Float16* xh  = wsh;                     // NX
  _Float16* wqh = xh  + (size_t)NX;        // NW
  _Float16* wkh = wqh + (size_t)NW;
  _Float16* wvh = wkh + (size_t)NW;
  _Float16* woh = wvh + (size_t)NW;
  _Float16* Qh  = woh + (size_t)NW;        // NX
  _Float16* Kh  = Qh  + (size_t)NX;        // NX
  _Float16* Vth = Kh  + (size_t)NX;        // NX  ([B][H][S])
  _Float16* Sc  = Vth + (size_t)NX;        // B*S*S
  _Float16* Ao  = Qh;                      // alias: Q dead after scores

  dim3 blk(256);

  // 1) conversions
  cvt_f32_f16_kernel<<<NX / 256, blk, 0, stream>>>(x,  xh,  NX);
  cvt_f32_f16_kernel<<<NW / 256, blk, 0, stream>>>(wq, wqh, NW);
  cvt_f32_f16_kernel<<<NW / 256, blk, 0, stream>>>(wk, wkh, NW);
  cvt_f32_f16_kernel<<<NW / 256, blk, 0, stream>>>(wv, wvh, NW);
  cvt_f32_f16_kernel<<<NW / 256, blk, 0, stream>>>(wo, woh, NW);

  // 2) Q, K, V projections (V stored transposed per batch)
  int proj_blocks = (M / 128) * (HH / 64); // 2048
  gemm_proj_kernel<<<proj_blocks, blk, 0, stream>>>(xh, wqh, bq, Qh, nullptr, 0);
  gemm_proj_kernel<<<proj_blocks, blk, 0, stream>>>(xh, wkh, bk, Kh, nullptr, 0);
  gemm_proj_kernel<<<proj_blocks, blk, 0, stream>>>(xh, wvh, bv, Vth, nullptr, 1);

  // 3) masked scaled scores
  dim3 sgrid((SS / 128) * (SS / 64), BB); // (512, 8)
  gemm_scores_kernel<<<sgrid, blk, 0, stream>>>(Qh, Kh, am, Sc);

  // 4) softmax in place
  dim3 smgrid(SS, BB);
  softmax_kernel<<<smgrid, blk, 0, stream>>>(Sc);

  // 5) probs @ V
  dim3 pvgrid((SS / 128) * (HH / 64), BB); // (256, 8)
  gemm_pv_kernel<<<pvgrid, blk, 0, stream>>>(Sc, Vth, Ao);

  // 6) output projection -> f32 d_out
  gemm_proj_kernel<<<proj_blocks, blk, 0, stream>>>(Ao, woh, bo, nullptr, out, 2);
}
